// MultiHeadAttention_10840497455679
// MI455X (gfx1250) — matrix operand order repaired and checked by reading, not compiled, not run on hardware
//
#include <hip/hip_runtime.h>
#include <hip/hip_bf16.h>

// ---------------------------------------------------------------------------
// MHA forward for MI455X (gfx1250, wave32, WMMA, bf16 compute / f32 accum).
//   K0: query/key/value + Wq/Wk/Wv/Wo  f32 -> bf16 once (bandwidth-bound);
//       all GEMM inner loops then do pure bf16 b128 loads -> WMMA.
//   K1: q/k/v = X W^T + b   (head-split bf16 output)
//   K2: softmax over HEAD axis (reference quirk) -> per-(q,k) normalization
//       across 16 heads via LDS; streamed over keys; O = P V  (WMMA)
//   K3: out = O Wo^T + bo  (bf16 WMMA, f32 out)
// ---------------------------------------------------------------------------

#define D_MODEL 1024
#define N_HEADS 16
#define D_K     64
#define BATCH   4
#define SEQ     2048
#define ROWS    (BATCH * SEQ)          // 8192 flattened (b,s) rows
#define EPITCH  33                     // padded LDS row pitch (bank-conflict free)

typedef __attribute__((ext_vector_type(16))) __bf16 v16bf;
typedef __attribute__((ext_vector_type(2)))  __bf16 v2bf;
typedef __attribute__((ext_vector_type(8)))  float  v8f;

// f32 -> bf16, round-half-up (1 add); tie-vs-RNE differences are far below
// bf16 quantization noise.  Single element: add + high-half extract.
__device__ __forceinline__ __bf16 f32_to_bf16(float f) {
    unsigned u = __builtin_bit_cast(unsigned, f) + 0x8000u;
    return __builtin_bit_cast(__bf16, (unsigned short)(u >> 16));
}

// Packed pair: 2 adds + one v_perm_b32 (pack the two high halves).
#if __has_builtin(__builtin_amdgcn_cvt_pk_bf16_f32)
__device__ __forceinline__ v2bf pk_bf16(float lo, float hi) {
    return __builtin_amdgcn_cvt_pk_bf16_f32(lo, hi);
}
#else
__device__ __forceinline__ v2bf pk_bf16(float lo, float hi) {
    unsigned ul = __builtin_bit_cast(unsigned, lo) + 0x8000u;
    unsigned uh = __builtin_bit_cast(unsigned, hi) + 0x8000u;
    // result bytes [1:0] = ul bytes [3:2], bytes [3:2] = uh bytes [3:2]
    unsigned p = __builtin_amdgcn_perm(uh, ul, 0x07060302u);
    return __builtin_bit_cast(v2bf, p);
}
#endif

// --- WMMA fragment loaders ---------------------------------------------------
// V_WMMA_F32_16X16X32_BF16 operand layouts (ISA 7.12.2, 16-bit matrices):
//   A (16x32, MxK): lane L holds row M=L&15; K = (L>>4)*8 + j for j=0..7 and
//                   K = 16 + (L>>4)*8 + (j-8) for j=8..15.
//   B (32x16, KxN): lane L holds col N=L&15; element j holds row
//                   K = (L>>4)*16 + j in these loaders (A likewise); ISA order is as for A.
//   C/D (16x16 f32): VGPR r, lane L -> (M = r + (L>>4)*8, N = L&15).

union abuild { v16bf v; v2bf h[8]; };

// A-fragment from a row-major bf16 matrix, leading dim ld (2 x b128 per lane).
__device__ __forceinline__ v16bf load_a_bf16(const __bf16* __restrict__ src,
                                             int ld, int row0, int k0) {
    int lane = threadIdx.x & 31;
    const __bf16* p = src + (row0 + (lane & 15)) * ld + k0 + ((lane >> 4) << 4);
    v16bf a;
#pragma unroll
    for (int j = 0; j < 8; ++j) a[j] = p[j];
#pragma unroll
    for (int j = 0; j < 8; ++j) a[8 + j] = p[8 + j];
    return a;
}

// B-fragment where B[k][n] is read from a *transposed* source: B[k][n] =
// src[(n0+n)*ld + k0 + k].  Used for W^T (weights stored [out,in]) and for
// K^T in QK^T (K stored [s,d], contraction over d).  32 contiguous bytes/lane.
__device__ __forceinline__ v16bf load_bT_bf16(const __bf16* __restrict__ src,
                                              int ld, int k0, int n0) {
    int lane = threadIdx.x & 31;
    const __bf16* p = src + (n0 + (lane & 15)) * ld + k0 + ((lane >> 4) << 4);
    v16bf b;
#pragma unroll
    for (int j = 0; j < 16; ++j) b[j] = p[j];
    return b;
}

// B-fragment from a row-major source: B[k][n] = src[(k0+k)*ld + n0+n].
// Used for V in P·V (contraction over key rows).
__device__ __forceinline__ v16bf load_b_bf16(const __bf16* __restrict__ src,
                                             int ld, int k0, int n0) {
    int lane = threadIdx.x & 31;
    int n  = n0 + (lane & 15);
    int kb = k0 + ((lane >> 4) << 4);
    v16bf b;
#pragma unroll
    for (int j = 0; j < 16; ++j) b[j] = src[(kb + j) * ld + n];
    return b;
}

__device__ __forceinline__ v8f wmma_bf16(v16bf a, v16bf b, v8f c) {
    return __builtin_amdgcn_wmma_f32_16x16x32_bf16(
        /*neg_a=*/false, a, /*neg_b=*/false, b,
        /*c_mod=*/(short)0, c, /*reuse_a=*/false, /*reuse_b=*/false);
}

// ---------------------------------------------------------------------------
// Kernel 0: elementwise f32 -> bf16.  4 elements / thread (b128 in, b64 out).
// ---------------------------------------------------------------------------
__global__ void __launch_bounds__(256)
cvt_f32_to_bf16(const float4* __restrict__ src, v2bf* __restrict__ dst, int n4) {
    int i = blockIdx.x * blockDim.x + threadIdx.x;
    if (i < n4) {
        float4 f = src[i];
        dst[2 * i]     = pk_bf16(f.x, f.y);
        dst[2 * i + 1] = pk_bf16(f.z, f.w);
    }
}

// ---------------------------------------------------------------------------
// Kernel 1: fused QKV projection.  One wave -> 32x64 output strip
// (2 A-frags + 4 B-frags -> 8 WMMAs per k-step; all operands bf16).
// Writes bf16 in head-split layout [B, H, S, Dk].
// ---------------------------------------------------------------------------
__global__ void __launch_bounds__(256)
mha_qkv_proj(const __bf16* __restrict__ qx, const __bf16* __restrict__ kx,
             const __bf16* __restrict__ vx,
             const __bf16* __restrict__ Wqb, const float* __restrict__ bq,
             const __bf16* __restrict__ Wkb, const float* __restrict__ bk,
             const __bf16* __restrict__ Wvb, const float* __restrict__ bv,
             __bf16* __restrict__ qh, __bf16* __restrict__ kh,
             __bf16* __restrict__ vh) {
    const int STRIPS = (ROWS / 32) * (D_MODEL / 64);      // 4096 per matrix
    int wave = (int)(blockIdx.x * (blockDim.x >> 5)) + (threadIdx.x >> 5);
    int mtx  = wave / STRIPS;
    int rem  = wave % STRIPS;
    int row0 = (rem >> 4) * 32;                           // 256 row tiles
    int n0   = (rem & 15) * 64;                           // 16 col strips

    const __bf16* X; const __bf16* W; const float* bias; __bf16* dst;
    if (mtx == 0)      { X = qx; W = Wqb; bias = bq; dst = qh; }
    else if (mtx == 1) { X = kx; W = Wkb; bias = bk; dst = kh; }
    else               { X = vx; W = Wvb; bias = bv; dst = vh; }

    v8f acc[2][4] = {{v8f{}, v8f{}, v8f{}, v8f{}}, {v8f{}, v8f{}, v8f{}, v8f{}}};
#pragma unroll 2
    for (int kk = 0; kk < D_MODEL; kk += 32) {
        v16bf a0 = load_a_bf16(X, D_MODEL, row0,      kk);
        v16bf a1 = load_a_bf16(X, D_MODEL, row0 + 16, kk);
#pragma unroll
        for (int t = 0; t < 4; ++t) {
            v16bf b = load_bT_bf16(W, D_MODEL, kk, n0 + t * 16);   // B = W^T
            acc[0][t] = wmma_bf16(a0, b, acc[0][t]);
            acc[1][t] = wmma_bf16(a1, b, acc[1][t]);
        }
    }

    int lane = threadIdx.x & 31;
    int nloc = lane & 15;
    int mhi  = (lane >> 4) << 3;
#pragma unroll
    for (int u = 0; u < 2; ++u) {
#pragma unroll
        for (int t = 0; t < 4; ++t) {
#pragma unroll
            for (int r = 0; r < 8; ++r) {
                int i = row0 + u * 16 + r + mhi;      // flattened (b,s)
                int j = n0 + t * 16 + nloc;           // model dim
                float val = acc[u][t][r] + bias[j];
                int bi = i >> 11, s = i & (SEQ - 1);
                int h = j >> 6, dk = j & (D_K - 1);
                dst[(((bi * N_HEADS) + h) * SEQ + s) * D_K + dk] = f32_to_bf16(val);
            }
        }
    }
}

// ---------------------------------------------------------------------------
// Kernel 2: attention with softmax over the HEAD axis.
// Block = 16 waves (one head each) x 32 lanes; one 16-query tile per block.
// Normalization across heads is purely per-(q,k) -> stream keys in tiles
// of 32 with no running max/sum; exp tiles exchanged via LDS (which also
// performs the C-layout -> A-layout shuffle for the P·V WMMA).
// ---------------------------------------------------------------------------
__global__ void __launch_bounds__(512)
mha_attention(const __bf16* __restrict__ qh, const __bf16* __restrict__ kh,
              const __bf16* __restrict__ vh, __bf16* __restrict__ combined) {
    __shared__ float lds_e[N_HEADS * 16 * EPITCH];   // exp(scores) [h][q][k<32]
    __shared__ float lds_rden[16 * EPITCH];          // 1 / sum-over-heads

    int b  = blockIdx.x >> 7;                        // SEQ/16 = 128 q-tiles
    int q0 = (blockIdx.x & 127) * 16;
    int h  = threadIdx.x >> 5;                       // head = wave id
    int lane = threadIdx.x & 31;

    const __bf16* Qh = qh + ((b * N_HEADS + h) * SEQ) * D_K;
    const __bf16* Kh = kh + ((b * N_HEADS + h) * SEQ) * D_K;
    const __bf16* Vh = vh + ((b * N_HEADS + h) * SEQ) * D_K;

    // Q fragments for this head's 16 query rows (Dk = 64 = 2 x K32), resident.
    v16bf aq0 = load_a_bf16(Qh, D_K, q0, 0);
    v16bf aq1 = load_a_bf16(Qh, D_K, q0, 32);

    v8f o[4] = {v8f{}, v8f{}, v8f{}, v8f{}};         // 16 x 64 f32 accumulator

    const float scale = 0.125f;                      // 1/sqrt(Dk)
    int nloc = lane & 15;
    int mhi  = (lane >> 4) << 3;

    for (int kb = 0; kb < SEQ; kb += 32) {
        // ---- S = Q K^T for two 16-wide key halves, exp to LDS ----
#pragma unroll
        for (int hf = 0; hf < 2; ++hf) {
            int col0 = kb + hf * 16;                 // key rows of this half
            v8f sc = v8f{};
            sc = wmma_bf16(aq0, load_bT_bf16(Kh, D_K, 0,  col0), sc);
            sc = wmma_bf16(aq1, load_bT_bf16(Kh, D_K, 32, col0), sc);
#pragma unroll
            for (int r = 0; r < 8; ++r) {
                int m  = r + mhi;
                int nl = hf * 16 + nloc;
                lds_e[(h * 16 + m) * EPITCH + nl] = __expf(sc[r] * scale);
            }
        }
        __syncthreads();

        // ---- reciprocal denominator: 512 threads x 1 (q,k) element ----
        {
            int q  = threadIdx.x >> 5;               // 0..15
            int kn = threadIdx.x & 31;               // 0..31
            float s = 0.f;
#pragma unroll
            for (int hh = 0; hh < N_HEADS; ++hh)
                s += lds_e[(hh * 16 + q) * EPITCH + kn];
            lds_rden[q * EPITCH + kn] = __builtin_amdgcn_rcpf(s);
        }
        __syncthreads();

        // ---- P fragment (A layout, 16x32 over keys) straight from LDS ----
        v16bf pfrag;
        {
            int m  = lane & 15;
            int kq = (lane >> 4) << 4;
            const float* er = &lds_e[(h * 16 + m) * EPITCH];
            const float* dr = &lds_rden[m * EPITCH];
            abuild pb;
#pragma unroll
            for (int j = 0; j < 8; j += 2) {
                int kkl = kq + j;
                pb.h[j >> 1] = pk_bf16(er[kkl]     * dr[kkl],
                                       er[kkl + 1] * dr[kkl + 1]);
            }
#pragma unroll
            for (int j = 0; j < 8; j += 2) {
                int kkl = 8 + kq + j;
                pb.h[(8 + j) >> 1] = pk_bf16(er[kkl]     * dr[kkl],
                                             er[kkl + 1] * dr[kkl + 1]);
            }
            pfrag = pb.v;
        }

        // ---- O += P (16x32) x V (32x64) ----
#pragma unroll
        for (int d = 0; d < 4; ++d)
            o[d] = wmma_bf16(pfrag, load_b_bf16(Vh, D_K, kb, d * 16), o[d]);

        __syncthreads();   // protect lds_e/lds_rden before next key tile
    }

    // ---- store combined[b, q, h*64 + d] (bf16) ----
#pragma unroll
    for (int d = 0; d < 4; ++d) {
#pragma unroll
        for (int r = 0; r < 8; ++r) {
            int m = r + mhi;
            int row = b * SEQ + q0 + m;
            int col = h * D_K + d * 16 + nloc;
            combined[row * D_MODEL + col] = f32_to_bf16(o[d][r]);
        }
    }
}

// ---------------------------------------------------------------------------
// Kernel 3: output projection  out = combined Wo^T + bo  (f32 out).
// One wave -> 32x64 strip, all-bf16 operand loads.
// ---------------------------------------------------------------------------
__global__ void __launch_bounds__(256)
mha_out_proj(const __bf16* __restrict__ combined,
             const __bf16* __restrict__ Wob, const float* __restrict__ bo,
             float* __restrict__ out) {
    int wave = (int)(blockIdx.x * (blockDim.x >> 5)) + (threadIdx.x >> 5);
    int row0 = (wave >> 4) * 32;
    int n0   = (wave & 15) * 64;

    v8f acc[2][4] = {{v8f{}, v8f{}, v8f{}, v8f{}}, {v8f{}, v8f{}, v8f{}, v8f{}}};
#pragma unroll 2
    for (int kk = 0; kk < D_MODEL; kk += 32) {
        v16bf a0 = load_a_bf16(combined, D_MODEL, row0,      kk);
        v16bf a1 = load_a_bf16(combined, D_MODEL, row0 + 16, kk);
#pragma unroll
        for (int t = 0; t < 4; ++t) {
            v16bf b = load_bT_bf16(Wob, D_MODEL, kk, n0 + t * 16);
            acc[0][t] = wmma_bf16(a0, b, acc[0][t]);
            acc[1][t] = wmma_bf16(a1, b, acc[1][t]);
        }
    }

    int lane = threadIdx.x & 31;
    int nloc = lane & 15;
    int mhi  = (lane >> 4) << 3;
#pragma unroll
    for (int u = 0; u < 2; ++u) {
#pragma unroll
        for (int t = 0; t < 4; ++t) {
#pragma unroll
            for (int r = 0; r < 8; ++r) {
                int i = row0 + u * 16 + r + mhi;
                int j = n0 + t * 16 + nloc;
                out[i * D_MODEL + j] = acc[u][t][r] + bo[j];
            }
        }
    }
}

// ---------------------------------------------------------------------------
extern "C" void kernel_launch(void* const* d_in, const int* in_sizes, int n_in,
                              void* d_out, int out_size, void* d_ws, size_t ws_size,
                              hipStream_t stream) {
    const float* query = (const float*)d_in[0];
    const float* key   = (const float*)d_in[1];
    const float* value = (const float*)d_in[2];
    const float* Wq = (const float*)d_in[3];  const float* bq = (const float*)d_in[4];
    const float* Wk = (const float*)d_in[5];  const float* bk = (const float*)d_in[6];
    const float* Wv = (const float*)d_in[7];  const float* bv = (const float*)d_in[8];
    const float* Wo = (const float*)d_in[9];  const float* bo = (const float*)d_in[10];
    float* out = (float*)d_out;

    // Workspace (bf16 elements):
    //   qx|kx|vx (8M each) | qh|kh|vh (8M each) | combined (8M) | 4 weights (1M)
    //   = 7*8M + 4*1M elems  ->  ~126 MB
    const size_t ACT_ELEMS = (size_t)ROWS * D_MODEL;                  // 8388608
    const size_t W_ELEMS   = (size_t)D_MODEL * D_MODEL;               // 1048576
    __bf16* wsb      = (__bf16*)d_ws;
    __bf16* qx       = wsb;
    __bf16* kx       = wsb + ACT_ELEMS;
    __bf16* vx       = wsb + 2 * ACT_ELEMS;
    __bf16* qh       = wsb + 3 * ACT_ELEMS;
    __bf16* kh       = wsb + 4 * ACT_ELEMS;
    __bf16* vh       = wsb + 5 * ACT_ELEMS;
    __bf16* combined = wsb + 6 * ACT_ELEMS;
    __bf16* Wqb      = wsb + 7 * ACT_ELEMS;
    __bf16* Wkb      = Wqb + W_ELEMS;
    __bf16* Wvb      = Wkb + W_ELEMS;
    __bf16* Wob      = Wvb + W_ELEMS;

    // K0: one-time f32 -> bf16 conversions (activations + weights).
    {
        int n4a = (int)(ACT_ELEMS / 4);              // 2097152 float4 per input
        int ba  = (n4a + 255) / 256;
        cvt_f32_to_bf16<<<ba, 256, 0, stream>>>((const float4*)query, (v2bf*)qx, n4a);
        cvt_f32_to_bf16<<<ba, 256, 0, stream>>>((const float4*)key,   (v2bf*)kx, n4a);
        cvt_f32_to_bf16<<<ba, 256, 0, stream>>>((const float4*)value, (v2bf*)vx, n4a);
        int n4w = (int)(W_ELEMS / 4);                // 262144 float4 per matrix
        int bw  = (n4w + 255) / 256;
        cvt_f32_to_bf16<<<bw, 256, 0, stream>>>((const float4*)Wq, (v2bf*)Wqb, n4w);
        cvt_f32_to_bf16<<<bw, 256, 0, stream>>>((const float4*)Wk, (v2bf*)Wkb, n4w);
        cvt_f32_to_bf16<<<bw, 256, 0, stream>>>((const float4*)Wv, (v2bf*)Wvb, n4w);
        cvt_f32_to_bf16<<<bw, 256, 0, stream>>>((const float4*)Wo, (v2bf*)Wob, n4w);
    }
    // K1: 3 matrices x 4096 strips of 32x64; 8 waves / 256-thread block.
    {
        int waves  = 3 * (ROWS / 32) * (D_MODEL / 64);
        int blocks = waves / 8;
        mha_qkv_proj<<<blocks, 256, 0, stream>>>(qx, kx, vx,
                                                 Wqb, bq, Wkb, bk, Wvb, bv,
                                                 qh, kh, vh);
    }
    // K2: one block per (batch, 16-query tile); 16 waves = 16 heads.
    {
        int blocks = BATCH * (SEQ / 16);             // 512
        mha_attention<<<blocks, 512, 0, stream>>>(qh, kh, vh, combined);
    }
    // K3: 4096 strips of 32x64; 8 waves / block.
    {
        int waves  = (ROWS / 32) * (D_MODEL / 64);
        int blocks = waves / 8;
        mha_out_proj<<<blocks, 256, 0, stream>>>(combined, Wob, bo, out);
    }
}
